// ShiftedWindowAttention_20899310863070
// MI455X (gfx1250) — compile-verified, matrix operand order checked by reading, not run on hardware
//
#include <hip/hip_runtime.h>
#include <hip/hip_bf16.h>

typedef __attribute__((ext_vector_type(16))) _Float16 v16h;
typedef __attribute__((ext_vector_type(8)))  _Float16 v8h;
typedef __attribute__((ext_vector_type(4)))  _Float16 v4h;
typedef __attribute__((ext_vector_type(8)))  float    v8f;

#define DIM    96
#define NH     3
#define HD     32
#define WS     7
#define SHIFT  3
#define IMG    224
#define NTOK   49
#define MPAD   64

#define XS_STRIDE 104   // halfs; 208B rows (16B aligned)
#define QK_STRIDE 200   // halfs; 400B rows
#define VT_STRIDE 72    // halfs; 144B rows
#define XS_OFF  0
#define QK_OFF  13312                 // 64*104*2
#define VT_OFF  (QK_OFF + 25600)     // 64*200*2 -> 38912
#define S_OFF   (VT_OFF + 13824)     // 96*72*2  -> 52736
#define LDS_BYTES (S_OFF + 49152)    // 3*64*64*4 -> 101888

// Fragment loader for 16-bit WMMA A/B where K is contiguous in memory.
// CDNA5 16-bit layout: lanes 0-15 hold kk {base..base+7, base+16..base+23},
// lanes 16-31 the same shifted by +8. Two contiguous 16B chunks per lane.
__device__ __forceinline__ v16h load_frag_k(const _Float16* base, int row,
                                            int stride, int k0, int lane) {
  const _Float16* p = base + row * stride + k0 + ((lane & 16) ? 8 : 0);
  v8h lo = *(const v8h*)p;
  v8h hi = *(const v8h*)(p + 16);
  return __builtin_shufflevector(lo, hi, 0,1,2,3,4,5,6,7,8,9,10,11,12,13,14,15);
}

__global__ void prep_weights(const float* __restrict__ wqkv,
                             const float* __restrict__ wproj,
                             _Float16* __restrict__ wqkvT,
                             _Float16* __restrict__ wprojT) {
  int i = blockIdx.x * blockDim.x + threadIdx.x;
  if (i < DIM * 3 * DIM) {                 // w_qkv [96][288] -> wqkvT [288][96] f16
    int k = i / (3 * DIM), n = i - k * (3 * DIM);
    wqkvT[n * DIM + k] = (_Float16)wqkv[i];
  }
  int j = i - DIM * 3 * DIM;
  if (j >= 0 && j < DIM * DIM) {           // w_proj [96][96] -> wprojT [96][96] f16
    int k = j / DIM, n = j - k * DIM;
    wprojT[n * DIM + k] = (_Float16)wproj[j];
  }
}

__global__ __launch_bounds__(288)
void swin_attn_kernel(const float* __restrict__ x,
                      const _Float16* __restrict__ wqkvT,
                      const float* __restrict__ bqkv,
                      const _Float16* __restrict__ wprojT,
                      const float* __restrict__ bproj,
                      float* __restrict__ out) {
  extern __shared__ __align__(16) char smem[];
  _Float16* Xs = (_Float16*)(smem + XS_OFF);   // window / later O, [64][104]
  _Float16* QK = (_Float16*)(smem + QK_OFF);   // Q(scaled)|K row-major [64][200]
  _Float16* Vt = (_Float16*)(smem + VT_OFF);   // V transposed [96 d][72 tok]
  float*    Sf = (float*)   (smem + S_OFF);    // scores  [3][64][64] f32
  _Float16* Pf = (_Float16*)(smem + S_OFF);    // probs   [3][64] rows, stride 128 halfs

  const int tid  = threadIdx.x;
  const int lane = tid & 31;
  const int wave = tid >> 5;
  const int hi16 = (lane & 16) ? 8 : 0;  // C/D tile row offset for upper lanes
  const int ln   = lane & 15;

  const int wid  = blockIdx.x;
  const int bimg = wid >> 10;
  const int wy   = (wid >> 5) & 31;
  const int wx   = wid & 31;

  // ---- Phase 0: gather shifted window -> LDS f16 (float4 loads), zero pad ----
  const float4* xv = (const float4*)x;
  for (int e = tid; e < MPAD * (DIM / 4); e += 288) {
    int row = e / (DIM / 4), c4 = e - row * (DIM / 4);
    v4h pk = {};
    if (row < NTOK) {
      int i = row / WS, j = row - (row / WS) * WS;
      int gh = (wy * WS + i + SHIFT) % IMG;
      int gw = (wx * WS + j + SHIFT) % IMG;
      float4 v = xv[((size_t)(bimg * IMG + gh) * IMG + gw) * (DIM / 4) + c4];
      pk[0] = (_Float16)v.x; pk[1] = (_Float16)v.y;
      pk[2] = (_Float16)v.z; pk[3] = (_Float16)v.w;
    }
    *(v4h*)(Xs + row * XS_STRIDE + c4 * 4) = pk;
  }
  __syncthreads();

  // ---- Phase 1: QKV gemm. 18 N-tiles over 9 waves, 4 M-tiles, K=96 ----
  const float qscale = 0.17677669529663687f;   // 1/sqrt(32), folded into Q
  for (int nt = wave * 2; nt < wave * 2 + 2; ++nt) {
    const int n0 = nt * 16;
    v16h bfrag[3];
#pragma unroll
    for (int ks = 0; ks < 3; ++ks)
      bfrag[ks] = load_frag_k(wqkvT, n0 + ln, DIM, ks * 32, lane);
    const float bias = bqkv[n0 + ln];
#pragma unroll
    for (int mt = 0; mt < 4; ++mt) {
      const int m0 = mt * 16;
      v8f acc = {};
#pragma unroll
      for (int ks = 0; ks < 3; ++ks) {
        v16h a = load_frag_k(Xs, m0 + ln, XS_STRIDE, ks * 32, lane);
        acc = __builtin_amdgcn_wmma_f32_16x16x32_f16(false, a, false, bfrag[ks],
                                                     (short)0, acc, false, false);
      }
      if (nt < 12) {                       // Q (scaled) and K -> row-major
        const float sc = (nt < 6) ? qscale : 1.0f;
#pragma unroll
        for (int r = 0; r < 8; ++r)
          QK[(m0 + r + hi16) * QK_STRIDE + n0 + ln] = (_Float16)((acc[r] + bias) * sc);
      } else {                             // V -> transposed [d][token]
        v8h pk;
#pragma unroll
        for (int r = 0; r < 8; ++r) pk[r] = (_Float16)(acc[r] + bias);
        *(v8h*)(Vt + (n0 - 2 * DIM + ln) * VT_STRIDE + m0 + hi16) = pk;
      }
    }
  }
  __syncthreads();

  // ---- Phase 2: S = Q @ K^T (48 tiles), mask cols >= 49 ----
  for (int t = wave; t < 48; t += 9) {
    const int head = t >> 4;
    const int rem  = t & 15;
    const int m0 = (rem >> 2) * 16;
    const int n0 = (rem & 3) * 16;
    v16h a = load_frag_k(QK, m0 + ln, QK_STRIDE, head * HD, lane);        // Q rows
    v16h b = load_frag_k(QK, n0 + ln, QK_STRIDE, DIM + head * HD, lane);  // K rows = B cols
    v8f acc = {};
    acc = __builtin_amdgcn_wmma_f32_16x16x32_f16(false, a, false, b,
                                                 (short)0, acc, false, false);
    const int ncol = n0 + ln;
    float* sh = Sf + head * (MPAD * MPAD);
#pragma unroll
    for (int r = 0; r < 8; ++r)
      sh[(m0 + r + hi16) * MPAD + ncol] = (ncol < NTOK) ? acc[r] : -1e30f;
  }
  __syncthreads();

  // ---- Phase 3: row softmax, write f16 probs in place (stride 128 halfs) ----
  if (tid < NH * MPAD) {
    const int head = tid >> 6, row = tid & 63;
    float*    srow = Sf + (head * MPAD + row) * MPAD;
    _Float16* prow = Pf + (head * MPAD + row) * (MPAD * 2);
    float mx = -1e30f;
    for (int n = 0; n < MPAD; ++n) mx = fmaxf(mx, srow[n]);
    float sum = 0.f;
    for (int n = 0; n < MPAD; ++n) {
      float e = exp2f((srow[n] - mx) * 1.44269504f);
      sum += e;
      prow[n] = (_Float16)e;               // clobbers only already-read bytes
    }
    float inv = 1.0f / sum;
    for (int n = 0; n < MPAD; ++n)
      prow[n] = (_Float16)((float)prow[n] * inv);
  }
  __syncthreads();

  // ---- Phase 4: O = P @ V (24 tiles, K=64), store row-major into Xs ----
  for (int t = wave; t < 24; t += 9) {
    const int head = t / 8;
    const int rem  = t % 8;
    const int m0 = (rem & 3) * 16;
    const int n0 = (rem >> 2) * 16;
    v8f acc = {};
#pragma unroll
    for (int ks = 0; ks < 2; ++ks) {
      v16h a = load_frag_k(Pf + head * (MPAD * MPAD * 2), m0 + ln, MPAD * 2, ks * 32, lane);
      v16h b = load_frag_k(Vt, head * HD + n0 + ln, VT_STRIDE, ks * 32, lane);
      acc = __builtin_amdgcn_wmma_f32_16x16x32_f16(false, a, false, b,
                                                   (short)0, acc, false, false);
    }
    const int col = head * HD + n0 + ln;
#pragma unroll
    for (int r = 0; r < 8; ++r)
      Xs[(m0 + r + hi16) * XS_STRIDE + col] = (_Float16)acc[r];
  }
  __syncthreads();

  // ---- Phase 5: Y = O @ Wproj + b, scatter-store with reverse roll ----
  for (int t = wave; t < 24; t += 9) {
    const int m0 = (t / 6) * 16;
    const int n0 = (t % 6) * 16;
    v8f acc = {};
#pragma unroll
    for (int ks = 0; ks < 3; ++ks) {
      v16h a = load_frag_k(Xs, m0 + ln, XS_STRIDE, ks * 32, lane);
      v16h b = load_frag_k(wprojT, n0 + ln, DIM, ks * 32, lane);
      acc = __builtin_amdgcn_wmma_f32_16x16x32_f16(false, a, false, b,
                                                   (short)0, acc, false, false);
    }
    const int col = n0 + ln;
    const float bias = bproj[col];
#pragma unroll
    for (int r = 0; r < 8; ++r) {
      const int m = m0 + r + hi16;
      if (m < NTOK) {
        int i = m / WS, j = m - (m / WS) * WS;
        int gh = (wy * WS + i + SHIFT) % IMG;
        int gw = (wx * WS + j + SHIFT) % IMG;
        out[(((size_t)bimg * IMG + gh) * IMG + gw) * DIM + col] = acc[r] + bias;
      }
    }
  }
}

extern "C" void kernel_launch(void* const* d_in, const int* in_sizes, int n_in,
                              void* d_out, int out_size, void* d_ws, size_t ws_size,
                              hipStream_t stream) {
  const float* x     = (const float*)d_in[0];
  const float* wqkv  = (const float*)d_in[1];
  const float* bqkv  = (const float*)d_in[2];
  const float* wproj = (const float*)d_in[3];
  const float* bproj = (const float*)d_in[4];
  float* out = (float*)d_out;

  _Float16* wqkvT  = (_Float16*)d_ws;                        // 288*96*2 = 55296 B
  _Float16* wprojT = (_Float16*)((char*)d_ws + 55296);       //  96*96*2 = 18432 B

  prep_weights<<<144, 256, 0, stream>>>(wqkv, wproj, wqkvT, wprojT);

  (void)hipFuncSetAttribute((const void*)swin_attn_kernel,
                            hipFuncAttributeMaxDynamicSharedMemorySize, LDS_BYTES);

  const int num_windows = 16 * 32 * 32; // 16384 windows
  swin_attn_kernel<<<num_windows, 288, LDS_BYTES, stream>>>(x, wqkvT, bqkv,
                                                            wprojT, bproj, out);
}
